// socialLstm_89824946029344
// MI455X (gfx1250) — matrix operand order repaired and checked by reading, not compiled, not run on hardware
//
#include <hip/hip_runtime.h>
#include <hip/hip_bf16.h>

typedef __attribute__((ext_vector_type(16))) _Float16 v16h;
typedef __attribute__((ext_vector_type(8)))  _Float16 v8h;
typedef __attribute__((ext_vector_type(8)))  float    v8f;

#define DEV static __device__ __forceinline__

// Blocks loop-invariant code motion across the recurrence loop WITHOUT
// destroying pointer provenance: an opaque zero offset is re-produced every
// iteration, so fragment loads can't be hoisted (no 1000-VGPR spills), yet the
// address is still derived from the global kernarg pointer and lowers to
// global_load_b128 (LOADcnt only) instead of flat_load (LOADcnt+DScnt).
#define OPAQUE_VAL(x) asm volatile("" : "+s"(x))

#if __has_builtin(__builtin_amdgcn_sched_group_barrier)
#define SGB(mask, size) __builtin_amdgcn_sched_group_barrier((mask), (size), 0)
#else
#define SGB(mask, size)
#endif

// D = A(16x32 f16) * B(32x16 f16) + C(16x16 f32)
DEV v8f wmma16x16x32(v16h a, v16h b, v8f c) {
  return __builtin_amdgcn_wmma_f32_16x16x32_f16(false, a, false, b, (short)0, c,
                                                false, false);
}

DEV float tanh_(float x) {
#if __has_builtin(__builtin_amdgcn_tanhf)
  return __builtin_amdgcn_tanhf(x);   // gfx1250 V_TANH_F32, branch-free
#else
  float e = __expf(2.f * x);
  return (e - 1.f) * __builtin_amdgcn_rcpf(e + 1.f);
#endif
}
DEV float sigmoid_(float x) { return __builtin_amdgcn_rcpf(1.f + __expf(-x)); }
DEV float lrelu_(float x) { return x >= 0.f ? x : 0.1f * x; }
// A-fragment element->K map (contiguous, matches B): lane kh = lane>>4, K = kh*16 + e
DEV int amapk(int e, int kh) { return kh * 16 + e; }

// ---------------------------------------------------------------------------
// Prep: f32 weights -> f16 [N][K]-major copies; fused biases.
// ---------------------------------------------------------------------------
__global__ __launch_bounds__(256) void prep_kernel(
    const float* __restrict__ Wih, const float* __restrict__ Whh,
    const float* __restrict__ bih, const float* __restrict__ bhh,
    const float* __restrict__ dWih, const float* __restrict__ dWhh,
    const float* __restrict__ dbih, const float* __restrict__ dbhh,
    _Float16* __restrict__ WihT, _Float16* __restrict__ WhhT,
    float* __restrict__ gbias, _Float16* __restrict__ dWihT,
    _Float16* __restrict__ dWhhT, float* __restrict__ dbias) {
  int i = blockIdx.x * 256 + threadIdx.x;
  if (i < 8192) {
    WihT[i] = (_Float16)Wih[i];                       // [256][32]
  } else if (i < 24576) {
    int j = i - 8192;  WhhT[j] = (_Float16)Whh[j];    // [256][64]
  } else if (i < 24832) {
    int j = i - 24576; gbias[j] = bih[j] + bhh[j];    // [256]
  } else if (i < 73984) {
    int j = i - 24832; dWihT[j] = (_Float16)dWih[j];  // [512][96]
  } else if (i < 139520) {
    int j = i - 73984; dWhhT[j] = (_Float16)dWhh[j];  // [512][128]
  } else if (i < 140032) {
    int j = i - 139520; dbias[j] = dbih[j] + dbhh[j]; // [512]
  }
}

// ---------------------------------------------------------------------------
// Encoder LSTM (hist + nbrs fused): 8448 wave-tiles of 16 rows.
// Per step: gates[16x256] = lrelu(x@Winp)@Wih.T + h@Whh.T + b  via WMMA.
// ---------------------------------------------------------------------------
__global__ __launch_bounds__(256) void enc_lstm_kernel(
    const float* __restrict__ hist, const float* __restrict__ nbrs,
    const float* __restrict__ Winp, const float* __restrict__ binp,
    const _Float16* __restrict__ WihT, const _Float16* __restrict__ WhhT,
    const float* __restrict__ gbias,
    _Float16* __restrict__ hist_h16, float* __restrict__ nbrs_h32) {
  __shared__ float sW[64];
  __shared__ float sb[32];
  __shared__ float sgb[256];
  __shared__ __align__(32) _Float16 sh[8][16 * 64];  // per-wave h tile [m][k]

  int tid = threadIdx.x;
  if (tid < 64) sW[tid] = Winp[tid];
  if (tid < 32) sb[tid] = binp[tid];
  sgb[tid] = gbias[tid];
  __syncthreads();

  int wave = tid >> 5, lane = tid & 31;
  long rowb = ((long)blockIdx.x * 8 + wave) * 16;
  bool isHist = rowb < 4096;
  int m = lane & 15, kh = lane >> 4;
  int n = m, kb = kh * 16;  // B-fragment: lane holds col n, K = kb..kb+15
  long row = rowb + m;

  v8f cacc[4];
  for (int i = 0; i < 4; ++i)
    for (int j = 0; j < 8; ++j) cacc[i][j] = 0.f;
  v16h hA0, hA1;
  for (int e = 0; e < 16; ++e) { hA0[e] = (_Float16)0.f; hA1[e] = (_Float16)0.f; }

  size_t opq = 0;

  for (int t = 0; t < 16; ++t) {
    OPAQUE_VAL(opq);                       // fresh "unknown zero" each step
    const _Float16* wih = WihT + opq;
    const _Float16* whh = WhhT + opq;

    // input embedding directly into A-fragment layout
    float x0, x1;
    if (isHist) {
      long o = ((long)t * 4096 + row) * 2;  x0 = hist[o]; x1 = hist[o + 1];
    } else {
      long o = ((long)t * 131072 + (row - 4096)) * 2;  x0 = nbrs[o]; x1 = nbrs[o + 1];
    }
    v16h ax;
#pragma unroll
    for (int e = 0; e < 16; ++e) {
      int k = amapk(e, kh);
      ax[e] = (_Float16)lrelu_(x0 * sW[2 * k] + x1 * sW[2 * k + 1] + sb[k]);
    }

#pragma unroll 1
    for (int ct = 0; ct < 4; ++ct) {  // 4 cell-channel tiles of 16
      v16h bfr[12];
#pragma unroll
      for (int g = 0; g < 4; ++g) {
        int col = (g * 4 + ct) * 16 + n;
        bfr[g * 3 + 0] = *(const v16h*)(wih + col * 32 + kb);
        bfr[g * 3 + 1] = *(const v16h*)(whh + col * 64 + kb);
        bfr[g * 3 + 2] = *(const v16h*)(whh + col * 64 + 32 + kb);
      }
      v8f acc[4];
#pragma unroll
      for (int g = 0; g < 4; ++g) {   // i, f, g, o gate tiles
        float bv = sgb[(g * 4 + ct) * 16 + n];
        v8f c;
#pragma unroll
        for (int j = 0; j < 8; ++j) c[j] = bv;
        c = wmma16x16x32(ax,  bfr[g * 3 + 0], c);
        c = wmma16x16x32(hA0, bfr[g * 3 + 1], c);
        c = wmma16x16x32(hA1, bfr[g * 3 + 2], c);
        acc[g] = c;
      }
      // Scheduler pipeline: keep ~4 fragments (8 b128 loads) in flight ahead
      // of the consuming WMMA so s_wait_loadcnt thresholds are non-zero.
      // 12 WMMA / 24 VMEM-reads per ct iteration:
      SGB(0x020, 8);                 // 8 loads ahead (4 fragments)
#pragma unroll
      for (int p = 0; p < 8; ++p) {  // then (1 WMMA, 2 loads) x 8
        SGB(0x008, 1);
        SGB(0x020, 2);
      }
#pragma unroll
      for (int p = 0; p < 4; ++p)    // trailing 4 WMMA drain
        SGB(0x008, 1);

#pragma unroll
      for (int j = 0; j < 8; ++j) {   // elementwise in C-fragment layout
        float iv = sigmoid_(acc[0][j]);
        float fv = sigmoid_(acc[1][j]);
        float gv = tanh_(acc[2][j]);
        float ov = sigmoid_(acc[3][j]);
        float cv = fv * cacc[ct][j] + iv * gv;
        cacc[ct][j] = cv;
        float hv = ov * tanh_(cv);
        sh[wave][(kh * 8 + j) * 64 + ct * 16 + n] = (_Float16)hv;
      }
    }
    // h: C layout -> A fragments via per-wave LDS tile
    v8h l0 = *(const v8h*)&sh[wave][m * 64 + kh * 16];
    v8h h0 = *(const v8h*)&sh[wave][m * 64 + kh * 16 + 8];
    v8h l1 = *(const v8h*)&sh[wave][m * 64 + 32 + kh * 16];
    v8h h1 = *(const v8h*)&sh[wave][m * 64 + 32 + kh * 16 + 8];
#pragma unroll
    for (int e = 0; e < 8; ++e) {
      hA0[e] = l0[e]; hA0[8 + e] = h0[e];
      hA1[e] = l1[e]; hA1[8 + e] = h1[e];
    }
  }

  for (int idx = lane; idx < 16 * 64; idx += 32) {
    int m2 = idx >> 6, k2 = idx & 63;
    _Float16 hv = sh[wave][idx];
    if (isHist) hist_h16[(rowb + m2) * 64 + k2] = hv;
    else        nbrs_h32[(rowb - 4096 + m2) * 64 + k2] = (float)hv;
  }
}

// ---------------------------------------------------------------------------
// Occupied-cell extraction: masked_scatter == neighbor 32b+j -> j-th True cell.
// ---------------------------------------------------------------------------
__global__ __launch_bounds__(256) void cells_kernel(
    const unsigned char* __restrict__ mask, int* __restrict__ cells) {
  int b = blockIdx.x * 256 + threadIdx.x;
  if (b >= 4096) return;
  int cnt = 0;
  for (int c = 0; c < 169; ++c) {
    if (mask[(long)b * 10816 + (long)c * 64]) {
      if (cnt < 32) cells[b * 32 + cnt] = c;
      ++cnt;
    }
  }
}

// hist_enc = lrelu(hist_h @ W_dym.T + b_dym)   [4096,32]
__global__ __launch_bounds__(256) void histenc_kernel(
    const _Float16* __restrict__ hist_h16, const float* __restrict__ Wdym,
    const float* __restrict__ bdym, float* __restrict__ hist_enc) {
  int i = blockIdx.x * 256 + threadIdx.x;  // 4096*32
  int b = i >> 5, s = i & 31;
  float acc = bdym[s];
  for (int k = 0; k < 64; ++k)
    acc += (float)hist_h16[b * 64 + k] * Wdym[s * 64 + k];
  hist_enc[i] = lrelu_(acc);
}

// Social pooling: gather-GEMM against L2-resident W_sp. soc permuted index
// for (e, gw, gh) is e*169 + gw*13 + gh.
__global__ __launch_bounds__(64) void social_kernel(
    const float* __restrict__ nbrs_h32, const int* __restrict__ cells,
    const float* __restrict__ Wsp, const float* __restrict__ bsp,
    float* __restrict__ social) {
  __shared__ float nh[2048];
  __shared__ int sc[32];
  int b = blockIdx.x, s = threadIdx.x;
  for (int i = s; i < 2048; i += 64) nh[i] = nbrs_h32[(long)b * 2048 + i];
  if (s < 32) sc[s] = cells[b * 32 + s];
  __syncthreads();
  float acc = bsp[s];
  for (int j = 0; j < 32; ++j) {
    int c = sc[j];
    int gh = c / 13, gw = c - gh * 13;
    const float* w = Wsp + (long)s * 10816 + gw * 13 + gh;
#pragma unroll 8
    for (int e = 0; e < 64; ++e) acc += nh[j * 64 + e] * w[e * 169];
  }
  social[b * 64 + s] = lrelu_(acc);
}

// ---------------------------------------------------------------------------
// Decoder input gates (enc is broadcast over T, so compute once):
// xg[4096,512] = [hist_enc|social](96) @ dWih.T + dbih + dbhh       (WMMA)
// ---------------------------------------------------------------------------
__global__ __launch_bounds__(256) void decin_kernel(
    const float* __restrict__ hist_enc, const float* __restrict__ social,
    const _Float16* __restrict__ dWihT, const float* __restrict__ dbias,
    float* __restrict__ xg) {
  int tid = threadIdx.x, wave = tid >> 5, lane = tid & 31;
  long rowb = ((long)blockIdx.x * 8 + wave) * 16;
  int m = lane & 15, kh = lane >> 4, n = m, kb = kh * 16;
  long row = rowb + m;
  v16h aE, aS0, aS1;
#pragma unroll
  for (int e = 0; e < 16; ++e) {
    int k = amapk(e, kh);
    aE[e]  = (_Float16)hist_enc[row * 32 + k];
    aS0[e] = (_Float16)social[row * 64 + k];
    aS1[e] = (_Float16)social[row * 64 + 32 + k];
  }
  for (int nt = 0; nt < 32; ++nt) {
    int col = nt * 16 + n;
    v16h b0 = *(const v16h*)(dWihT + col * 96 + kb);
    v16h b1 = *(const v16h*)(dWihT + col * 96 + 32 + kb);
    v16h b2 = *(const v16h*)(dWihT + col * 96 + 64 + kb);
    float bv = dbias[col];
    v8f c;
#pragma unroll
    for (int j = 0; j < 8; ++j) c[j] = bv;
    c = wmma16x16x32(aE,  b0, c);
    c = wmma16x16x32(aS0, b1, c);
    c = wmma16x16x32(aS1, b2, c);
#pragma unroll
    for (int j = 0; j < 8; ++j)
      xg[(rowb + kh * 8 + j) * 512 + col] = c[j];
  }
}

// ---------------------------------------------------------------------------
// Decoder LSTM (return_seq): per step gates = xg + h@dWhh.T, h written per t.
// ---------------------------------------------------------------------------
__global__ __launch_bounds__(256) void dec_lstm_kernel(
    const float* __restrict__ xg, const _Float16* __restrict__ dWhhT,
    _Float16* __restrict__ dech) {
  __shared__ __align__(32) _Float16 sh[8][16 * 128];
  int tid = threadIdx.x, wave = tid >> 5, lane = tid & 31;
  long rowb = ((long)blockIdx.x * 8 + wave) * 16;
  int m = lane & 15, kh = lane >> 4, n = m, kb = kh * 16;

  v8f cacc[8];
  for (int i = 0; i < 8; ++i)
    for (int j = 0; j < 8; ++j) cacc[i][j] = 0.f;
  v16h hA[4];
  for (int f = 0; f < 4; ++f)
    for (int e = 0; e < 16; ++e) hA[f][e] = (_Float16)0.f;

  size_t opq = 0;

  for (int t = 0; t < 16; ++t) {
    OPAQUE_VAL(opq);                       // fresh "unknown zero" each step
    const _Float16* whh = dWhhT + opq;
    const float* xgp = xg + opq;
#pragma unroll 1
    for (int ct = 0; ct < 8; ++ct) {  // 8 cell-channel tiles
      // Batch all 16 weight fragments + 32 xg scalars of this tile first.
      v16h bfr[16];
      v8f cin[4];
#pragma unroll
      for (int g = 0; g < 4; ++g) {
        int col = (g * 8 + ct) * 16 + n;
#pragma unroll
        for (int f = 0; f < 4; ++f)
          bfr[g * 4 + f] = *(const v16h*)(whh + col * 128 + f * 32 + kb);
#pragma unroll
        for (int j = 0; j < 8; ++j)
          cin[g][j] = xgp[(rowb + kh * 8 + j) * 512 + col];
      }
      v8f acc[4];
#pragma unroll
      for (int g = 0; g < 4; ++g) {
        v8f c = cin[g];
#pragma unroll
        for (int f = 0; f < 4; ++f)
          c = wmma16x16x32(hA[f], bfr[g * 4 + f], c);
        acc[g] = c;
      }
      // Pipeline hint: 16 WMMA, 64 VMEM reads (32 weight b128 + 32 xg b32):
      // run ~24 loads ahead, then 2 loads per WMMA (trailing 8 unconstrained).
      SGB(0x020, 24);
#pragma unroll
      for (int p = 0; p < 16; ++p) {
        SGB(0x008, 1);
        SGB(0x020, 2);
      }

#pragma unroll
      for (int j = 0; j < 8; ++j) {
        float iv = sigmoid_(acc[0][j]);
        float fv = sigmoid_(acc[1][j]);
        float gv = tanh_(acc[2][j]);
        float ov = sigmoid_(acc[3][j]);
        float cv = fv * cacc[ct][j] + iv * gv;
        cacc[ct][j] = cv;
        _Float16 h16 = (_Float16)(ov * tanh_(cv));
        int mr = kh * 8 + j;
        sh[wave][mr * 128 + ct * 16 + n] = h16;
        dech[((long)t * 4096 + rowb + mr) * 128 + ct * 16 + n] = h16;
      }
    }
#pragma unroll
    for (int f = 0; f < 4; ++f) {
      v8h lo = *(const v8h*)&sh[wave][m * 128 + f * 32 + kh * 16];
      v8h hi = *(const v8h*)&sh[wave][m * 128 + f * 32 + kh * 16 + 8];
#pragma unroll
      for (int e = 0; e < 8; ++e) { hA[f][e] = lo[e]; hA[f][8 + e] = hi[e]; }
    }
  }
}

// out[b,t,:] = act(dec_h @ W_out.T + b_out)
__global__ __launch_bounds__(256) void out_kernel(
    const _Float16* __restrict__ dech, const float* __restrict__ Wout,
    const float* __restrict__ bout, float* __restrict__ out) {
  int i = blockIdx.x * 256 + threadIdx.x;  // 4096*16
  int b = i >> 4, t = i & 15;
  float acc[5];
#pragma unroll
  for (int s = 0; s < 5; ++s) acc[s] = bout[s];
  const _Float16* h = dech + ((long)t * 4096 + b) * 128;
  for (int k = 0; k < 128; ++k) {
    float hv = (float)h[k];
#pragma unroll
    for (int s = 0; s < 5; ++s) acc[s] += hv * Wout[s * 128 + k];
  }
  float* o = out + (long)i * 5;
  o[0] = acc[0];
  o[1] = acc[1];
  o[2] = __expf(acc[2]);
  o[3] = __expf(acc[3]);
  o[4] = tanh_(acc[4]);
}

extern "C" void kernel_launch(void* const* d_in, const int* in_sizes, int n_in,
                              void* d_out, int out_size, void* d_ws, size_t ws_size,
                              hipStream_t stream) {
  (void)in_sizes; (void)n_in; (void)out_size; (void)ws_size;
  const float* hist = (const float*)d_in[0];
  const float* nbrs = (const float*)d_in[1];
  const float* Winp = (const float*)d_in[2];
  const float* binp = (const float*)d_in[3];
  const float* Wih  = (const float*)d_in[4];
  const float* Whh  = (const float*)d_in[5];
  const float* bih  = (const float*)d_in[6];
  const float* bhh  = (const float*)d_in[7];
  const float* Wdym = (const float*)d_in[8];
  const float* bdym = (const float*)d_in[9];
  const float* Wsp  = (const float*)d_in[10];
  const float* bsp  = (const float*)d_in[11];
  const float* dWih = (const float*)d_in[12];
  const float* dWhh = (const float*)d_in[13];
  const float* dbih = (const float*)d_in[14];
  const float* dbhh = (const float*)d_in[15];
  const float* Wout = (const float*)d_in[16];
  const float* bout = (const float*)d_in[17];
  const unsigned char* mask = (const unsigned char*)d_in[18];
  float* out = (float*)d_out;

  char* ws = (char*)d_ws;
  size_t off = 0;
  auto alloc = [&](size_t bytes) {
    char* p = ws + off;
    off = (off + bytes + 255) & ~(size_t)255;
    return p;
  };
  _Float16* WihT   = (_Float16*)alloc(8192  * sizeof(_Float16)); // [256][32]
  _Float16* WhhT   = (_Float16*)alloc(16384 * sizeof(_Float16)); // [256][64]
  float*    gbias  = (float*)   alloc(256   * sizeof(float));
  _Float16* dWihT  = (_Float16*)alloc(49152 * sizeof(_Float16)); // [512][96]
  _Float16* dWhhT  = (_Float16*)alloc(65536 * sizeof(_Float16)); // [512][128]
  float*    dbias  = (float*)   alloc(512   * sizeof(float));
  _Float16* histH  = (_Float16*)alloc((size_t)4096 * 64 * sizeof(_Float16));
  float*    nbrsH  = (float*)   alloc((size_t)131072 * 64 * sizeof(float));
  float*    histE  = (float*)   alloc((size_t)4096 * 32 * sizeof(float));
  float*    social = (float*)   alloc((size_t)4096 * 64 * sizeof(float));
  int*      cells  = (int*)     alloc((size_t)4096 * 32 * sizeof(int));
  float*    xgdec  = (float*)   alloc((size_t)4096 * 512 * sizeof(float));
  _Float16* dech   = (_Float16*)alloc((size_t)16 * 4096 * 128 * sizeof(_Float16));

  prep_kernel<<<547, 256, 0, stream>>>(Wih, Whh, bih, bhh, dWih, dWhh, dbih,
                                       dbhh, WihT, WhhT, gbias, dWihT, dWhhT,
                                       dbias);
  // (4096 + 131072) rows / 16 per wave / 8 waves per block = 1056 blocks
  enc_lstm_kernel<<<1056, 256, 0, stream>>>(hist, nbrs, Winp, binp, WihT, WhhT,
                                            gbias, histH, nbrsH);
  cells_kernel<<<16, 256, 0, stream>>>(mask, cells);
  histenc_kernel<<<512, 256, 0, stream>>>(histH, Wdym, bdym, histE);
  social_kernel<<<4096, 64, 0, stream>>>(nbrsH, cells, Wsp, bsp, social);
  decin_kernel<<<32, 256, 0, stream>>>(histE, social, dWihT, dbias, xgdec);
  dec_lstm_kernel<<<32, 256, 0, stream>>>(xgdec, dWhhT, dech);
  out_kernel<<<256, 256, 0, stream>>>(dech, Wout, bout, out);
}
